// RingAttention_53231824666815
// MI455X (gfx1250) — matrix operand order repaired and checked by reading, not compiled, not run on hardware
//
#include <hip/hip_runtime.h>
#include <hip/hip_bf16.h>

// Flash-attention (causal) for B=2,H=16,S=2048,D=128 on gfx1250 (MI455X).
// Transposed-score formulation: S^T = K*Q^T, O^T = V^T*P^T, so softmax state
// is per-lane scalar (row = lane) and P never round-trips through LDS.
// f16 WMMA matmuls, fp32 accumulation + fp32 online softmax.
// Double-buffered cooperative K/V staging in LDS (f16), global_prefetch ahead.

typedef __attribute__((ext_vector_type(16))) _Float16 v16h;
typedef __attribute__((ext_vector_type(8)))  _Float16 v8h;
typedef __attribute__((ext_vector_type(2)))  _Float16 v2h;
typedef __attribute__((ext_vector_type(8)))  float    v8f;
typedef __attribute__((ext_vector_type(8)))  unsigned v8u;

#define S_DIM   2048
#define D_DIM   128
#define BLOCK_M 64
#define WAVES   4
#define NTHR    128
#define KV_TILE 32
#define KPAD    8
#define KSTRIDE (D_DIM + KPAD)      // 136 halves = 272B row stride (16B multiple)
#define VPAD    8
#define VSTRIDE (KV_TILE + VPAD)    // 40 halves = 80B row stride (16B multiple)

__global__ __launch_bounds__(NTHR)
void fa_causal_wmma_kernel(const float* __restrict__ Q,
                           const float* __restrict__ K,
                           const float* __restrict__ V,
                           float* __restrict__ O) {
  // double-buffered K (row-major keys x dims) and V (transposed dims x keys), f16
  __shared__ __align__(16) _Float16 sK [2][KV_TILE][KSTRIDE];
  __shared__ __align__(16) _Float16 sVt[2][D_DIM][VSTRIDE];

  const int head = blockIdx.y;              // flat b*H + h
  const int m0   = blockIdx.x * BLOCK_M;    // first query row of this WG
  const int tid  = threadIdx.x;
  const int wave = tid >> 5;
  const int lane = tid & 31;
  const int ln   = lane & 15;               // N sub-index within half-wave
  const int hi   = lane >> 4;               // half-wave select
  const int qr0  = m0 + wave * 16;          // first query row of this wave
  const int qrow = qr0 + ln;                // THIS lane's query row (transposed form)

  const float scale = 0.08838834764831845f; // 1/sqrt(128)

  const float* Qh = Q + (size_t)head * S_DIM * D_DIM;
  const float* Kh = K + (size_t)head * S_DIM * D_DIM;
  const float* Vh = V + (size_t)head * S_DIM * D_DIM;
  float*       Oh = O + (size_t)head * S_DIM * D_DIM;

  // cooperative staging: 128 threads cover 32 keys x 128 dims (32 f32 each)
  const int skrow = tid >> 2;               // 0..31 key within tile
  const int sd0   = (tid & 3) * 32;         // dim chunk base

  auto stage = [&](int buf, int t0) {
    const float* kr = Kh + (size_t)(t0 + skrow) * D_DIM + sd0;
    const float* vr = Vh + (size_t)(t0 + skrow) * D_DIM + sd0;
    #pragma unroll
    for (int ch = 0; ch < 4; ++ch) {        // 8 floats per chunk
      const float4 ka = *(const float4*)(kr + 8 * ch);
      const float4 kb = *(const float4*)(kr + 8 * ch + 4);
      v8h h;
      h[0] = (_Float16)ka.x; h[1] = (_Float16)ka.y;
      h[2] = (_Float16)ka.z; h[3] = (_Float16)ka.w;
      h[4] = (_Float16)kb.x; h[5] = (_Float16)kb.y;
      h[6] = (_Float16)kb.z; h[7] = (_Float16)kb.w;
      *(v8h*)&sK[buf][skrow][sd0 + 8 * ch] = h;   // contiguous b128 store
      const float4 va = *(const float4*)(vr + 8 * ch);
      const float4 vb = *(const float4*)(vr + 8 * ch + 4);
      const int d = sd0 + 8 * ch;
      sVt[buf][d + 0][skrow] = (_Float16)va.x;
      sVt[buf][d + 1][skrow] = (_Float16)va.y;
      sVt[buf][d + 2][skrow] = (_Float16)va.z;
      sVt[buf][d + 3][skrow] = (_Float16)va.w;
      sVt[buf][d + 4][skrow] = (_Float16)vb.x;
      sVt[buf][d + 5][skrow] = (_Float16)vb.y;
      sVt[buf][d + 6][skrow] = (_Float16)vb.z;
      sVt[buf][d + 7][skrow] = (_Float16)vb.w;
    }
  };

  // ---- Q^T in WMMA B-layout (32x16 per chunk), pre-scaled, f16.
  // B-layout: lane ln = column (query row), half-wave hi holds K(=dim) 16*hi+0..15.
  v16h qb[4];
  {
    const float* qr = Qh + (size_t)qrow * D_DIM + 16 * hi;
    #pragma unroll
    for (int c = 0; c < 4; ++c)
      #pragma unroll
      for (int i = 0; i < 16; ++i)
        qb[c][i] = (_Float16)(qr[32 * c + i] * scale);
  }

  v8f accT[8];            // O^T accumulators: tile c holds dims 16c+(j+8*hi), row=ln
  #pragma unroll
  for (int c = 0; c < 8; ++c) accT[c] = (v8f){};
  float mrow = -1e30f, lrow = 0.f;   // per-lane online-softmax state (row = qrow)

  const int nblocks = (m0 + BLOCK_M) / KV_TILE;  // causal: keys <= m0+63, >= 2

  stage(0, 0);
  __syncthreads();

  for (int blk = 0; blk < nblocks; ++blk) {
    const int t0  = blk * KV_TILE;
    const int cur = blk & 1;

    // stage next tile into the other buffer (overlaps with compute below)
    if (blk + 1 < nblocks) stage(1 - cur, t0 + KV_TILE);
    // prefetch tile-after-next into L2 (global_prefetch_b8)
    if (blk + 2 < nblocks) {
      const size_t poff = (size_t)(t0 + 2 * KV_TILE) * D_DIM + (size_t)tid * 32;
      __builtin_prefetch(Kh + poff, 0, 3);
      __builtin_prefetch(Vh + poff, 0, 3);
    }

    if (t0 <= qr0 + 15) {                  // causal: this wave needs this block
      // ---- S^T = K * Q^T : two 16x16 tiles (keys t0..+15, t0+16..+31)
      // A = K subtile in A-layout straight from sK:
      //   lane ln = key, halves 0..15 = dims 32c+16*hi+0..15
      v8f stile[2];
      #pragma unroll
      for (int t = 0; t < 2; ++t) {
        v8f c8 = (v8f){};
        #pragma unroll
        for (int c = 0; c < 4; ++c) {
          const v8h* klo = (const v8h*)&sK[cur][16 * t + ln][32 * c + 16 * hi];
          const v8h* khi = (const v8h*)&sK[cur][16 * t + ln][32 * c + 16 * hi + 8];
          const v16h ka = __builtin_shufflevector(klo[0], khi[0],
              0, 1, 2, 3, 4, 5, 6, 7, 8, 9, 10, 11, 12, 13, 14, 15);
          c8 = __builtin_amdgcn_wmma_f32_16x16x32_f16(
                  false, ka, false, qb[c], (short)0, c8, false, false);
        }
        stile[t] = c8;
      }

      // ---- causal mask (single partial block per wave) ----
      // S^T C-layout: lane's element (t,j) = score(key t0+16t+j+8hi, row qrow)
      if (t0 + KV_TILE - 1 > qr0) {
        #pragma unroll
        for (int t = 0; t < 2; ++t)
          #pragma unroll
          for (int j = 0; j < 8; ++j)
            if (t0 + 16 * t + j + 8 * hi > qrow) stile[t][j] = -1e30f;
      }

      // ---- per-lane online softmax (row = qrow) ----
      float mx = -1e30f;
      #pragma unroll
      for (int t = 0; t < 2; ++t)
        #pragma unroll
        for (int j = 0; j < 8; ++j)
          mx = fmaxf(mx, stile[t][j]);
      mx = fmaxf(mx, __shfl_xor(mx, 16, 32));     // other half-wave's keys
      const float mnew = fmaxf(mrow, mx);
      const float f    = __expf(mrow - mnew);
      float p[2][8];
      float ps = 0.f;
      #pragma unroll
      for (int t = 0; t < 2; ++t)
        #pragma unroll
        for (int j = 0; j < 8; ++j) {
          p[t][j] = __expf(stile[t][j] - mnew);
          ps += p[t][j];
        }
      ps += __shfl_xor(ps, 16, 32);
      lrow = lrow * f + ps;
      mrow = mnew;

      // rescale running output (f is a per-lane scalar: pure v_mul)
      #pragma unroll
      for (int c = 0; c < 8; ++c)
        #pragma unroll
        for (int j = 0; j < 8; ++j)
          accT[c][j] *= f;

      // ---- build P^T in B-layout via one cross-half exchange ----
      // own packed halves: subtile t, dword jj = (p[t][2jj], p[t][2jj+1])
      unsigned a0[4], a1[4], b0[4], b1[4];
      #pragma unroll
      for (int jj = 0; jj < 4; ++jj) {
        v2h h0; h0[0] = (_Float16)p[0][2 * jj]; h0[1] = (_Float16)p[0][2 * jj + 1];
        v2h h1; h1[0] = (_Float16)p[1][2 * jj]; h1[1] = (_Float16)p[1][2 * jj + 1];
        a0[jj] = __builtin_bit_cast(unsigned, h0);
        a1[jj] = __builtin_bit_cast(unsigned, h1);
        b0[jj] = (unsigned)__shfl_xor((int)a0[jj], 16, 32);
        b1[jj] = (unsigned)__shfl_xor((int)a1[jj], 16, 32);
      }
      // B-layout: halves 0..15 = keys 16*hi + 0..15 for column ln
      //   hi=0: keys 0..15  = own sub0 (0..7)  | partner sub0 (8..15)
      //   hi=1: keys 16..31 = partner sub1 (16..23) | own sub1 (24..31)
      v8u pk;
      #pragma unroll
      for (int jj = 0; jj < 4; ++jj) {
        pk[jj]     = hi ? b1[jj] : a0[jj];
        pk[4 + jj] = hi ? a1[jj] : b0[jj];
      }
      const v16h pb = __builtin_bit_cast(v16h, pk);

      // ---- O^T += V^T * P^T : 8 dim tiles, A = V^T straight from sVt ----
      #pragma unroll
      for (int c = 0; c < 8; ++c) {
        const v8h* vlo = (const v8h*)&sVt[cur][16 * c + ln][16 * hi];
        const v8h* vhi = (const v8h*)&sVt[cur][16 * c + ln][16 * hi + 8];
        const v16h va = __builtin_shufflevector(vlo[0], vhi[0],
            0, 1, 2, 3, 4, 5, 6, 7, 8, 9, 10, 11, 12, 13, 14, 15);
        accT[c] = __builtin_amdgcn_wmma_f32_16x16x32_f16(
                    false, va, false, pb, (short)0, accT[c], false, false);
      }
    }

    __syncthreads();   // staging of next buffer complete; prior reads done
  }

  // ---- epilogue: O[qrow][16c+8hi + 0..7] = accT[c][0..7] / l  (b128 stores)
  const float invl = 1.0f / lrow;
  float* orow = Oh + (size_t)qrow * D_DIM;
  #pragma unroll
  for (int c = 0; c < 8; ++c) {
    float4 lo, hi4;
    lo.x  = accT[c][0] * invl; lo.y  = accT[c][1] * invl;
    lo.z  = accT[c][2] * invl; lo.w  = accT[c][3] * invl;
    hi4.x = accT[c][4] * invl; hi4.y = accT[c][5] * invl;
    hi4.z = accT[c][6] * invl; hi4.w = accT[c][7] * invl;
    *(float4*)(orow + 16 * c + 8 * hi)     = lo;
    *(float4*)(orow + 16 * c + 8 * hi + 4) = hi4;
  }
}

extern "C" void kernel_launch(void* const* d_in, const int* in_sizes, int n_in,
                              void* d_out, int out_size, void* d_ws, size_t ws_size,
                              hipStream_t stream) {
  const float* q = (const float*)d_in[0];
  const float* k = (const float*)d_in[1];
  const float* v = (const float*)d_in[2];
  float* out = (float*)d_out;

  const int BH = 2 * 16;                       // B*H
  dim3 grid(S_DIM / BLOCK_M, BH);
  dim3 block(NTHR);
  fa_causal_wmma_kernel<<<grid, block, 0, stream>>>(q, k, v, out);
}
